// GeoAttnUpConv_73512660238921
// MI455X (gfx1250) — compile-verified, matrix operand order checked by reading, not run on hardware
//
#include <hip/hip_runtime.h>
#include <hip/hip_bf16.h>
#include <math.h>

// ---- problem constants (match reference) ----
#define BB   4
#define CIN  128
#define NN   4096
#define KNB  16      // K neighbors
#define HH   128     // edge-mlp hidden
#define COUT 256
#define SEH  64
#define NC   16      // N / QUERY_CHUNK

typedef float    v2f  __attribute__((ext_vector_type(2)));
typedef float    v8f  __attribute__((ext_vector_type(8)));
typedef _Float16 v16h __attribute__((ext_vector_type(16)));

union V16H { v16h v; uint4 q[2]; };

static __device__ inline v8f zero8() {
    v8f z;
#pragma unroll
    for (int i = 0; i < 8; ++i) z[i] = 0.0f;
    return z;
}

// async copy of one 16-byte chunk: global -> LDS (ASYNCcnt-tracked, CDNA5)
static __device__ inline void async_b128(unsigned lds_addr, unsigned long long gaddr) {
    asm volatile("global_load_async_to_lds_b128 %0, %1, off"
                 :: "v"(lds_addr), "v"(gaddr) : "memory");
}
static __device__ inline void wait_async0() {
    asm volatile("s_wait_asynccnt 0x0" ::: "memory");
}

// ---------------------------------------------------------------------------
// K0: transpose x (B,Cin,N) -> xt (B*N, Cin) f32 + xth f16, and row norms sq
// ---------------------------------------------------------------------------
__global__ __launch_bounds__(256) void k_prep_x(const float* __restrict__ x,
                                                float* __restrict__ xt,
                                                _Float16* __restrict__ xth,
                                                float* __restrict__ sq) {
    int id = blockIdx.x * 256 + threadIdx.x;       // 0 .. B*N-1
    int b = id >> 12, n = id & (NN - 1);
    const float* px = x + (size_t)b * CIN * NN + n;
    float*    pt = xt  + (size_t)id * CIN;
    _Float16* ph = xth + (size_t)id * CIN;
    float acc = 0.0f;
#pragma unroll 4
    for (int c = 0; c < CIN; ++c) {
        float v = px[(size_t)c * NN];
        acc += v * v;
        pt[c] = v;
        ph[c] = (_Float16)v;
    }
    sq[id] = acc;
}

// ---------------------------------------------------------------------------
// K0b: f16 weight copies + folded BN scale/bias
// ---------------------------------------------------------------------------
__global__ __launch_bounds__(256) void k_prep_w(
    const float* We, const float* Wf, const float* Wr,
    const float* g1, const float* bb1, const float* m1, const float* v1,
    const float* g2, const float* bb2, const float* m2, const float* v2,
    const float* g3, const float* bb3, const float* m3, const float* v3,
    _Float16* WeH, _Float16* WfH, _Float16* WrH,
    float* s1, float* b1, float* s2, float* b2, float* s3, float* b3) {
    int id = blockIdx.x * 256 + threadIdx.x;       // 0 .. 32767
    if (id < HH * 2 * CIN) WeH[id] = (_Float16)We[id];
    if (id < COUT * HH)    WfH[id] = (_Float16)Wf[id];
    if (id < COUT * CIN)   WrH[id] = (_Float16)Wr[id];
    if (id < HH) {
        float s = g1[id] * rsqrtf(v1[id] + 1e-5f);
        s1[id] = s;  b1[id] = bb1[id] - m1[id] * s;
    }
    if (id < COUT) {
        float s = g2[id] * rsqrtf(v2[id] + 1e-5f);
        s2[id] = s;  b2[id] = bb2[id] - m2[id] * s;
        float t = g3[id] * rsqrtf(v3[id] + 1e-5f);
        s3[id] = t;  b3[id] = bb3[id] - m3[id] * t;
    }
}

// ---------------------------------------------------------------------------
// K1: KNN via fp32 WMMA gram tiles.
//   Block = 4 waves; each wave owns a 16-row tile (64 rows per block).
//   Column panels (16 cols x 128 ch = 8 KB contiguous in xt) are async-DMA'd
//   into LDS, double buffered, and shared by all 4 waves.
// ---------------------------------------------------------------------------
__global__ __launch_bounds__(128) void k_knn(const float* __restrict__ xt,
                                             const float* __restrict__ sq,
                                             int* __restrict__ knn) {
    __shared__ float panel[2][16 * CIN];            // 2 x 8 KB
    __shared__ float tile[4][256];                  // per-wave distance tile

    int tid = threadIdx.x;
    int wave = tid >> 5;
    int l = tid & 31, half = l >> 4, lm = l & 15;
    int b = blockIdx.y;
    int rowbase = blockIdx.x * 64 + wave * 16;

    // A fragments: lane l holds M=lm, K = 4*kk + 2*half + {0,1}
    const float* rowp = xt + ((size_t)(b * NN + rowbase + lm)) * CIN;
    v2f a[32];
#pragma unroll
    for (int kk = 0; kk < 32; ++kk)
        a[kk] = *(const v2f*)(rowp + 4 * kk + 2 * half);

    float sqa[8];
#pragma unroll
    for (int r = 0; r < 8; ++r) sqa[r] = sq[b * NN + rowbase + r + 8 * half];

    float bd[16]; int bi[16];
#pragma unroll
    for (int s = 0; s < 16; ++s) { bd[s] = 3.0e38f; bi[s] = 0; }
    float wmax = 3.0e38f; int wpos = 0;

    const float* pbase = xt + (size_t)b * NN * CIN;
    unsigned ldsb[2];
    ldsb[0] = (unsigned)(unsigned long long)(void*)&panel[0][0];
    ldsb[1] = (unsigned)(unsigned long long)(void*)&panel[1][0];

    // prefetch panel for colbase = 0 into buffer 0
#pragma unroll
    for (int i = 0; i < 4; ++i)
        async_b128(ldsb[0] + tid * 16 + i * 2048,
                   (unsigned long long)pbase + tid * 16 + i * 2048);
    wait_async0();
    __syncthreads();

    for (int it = 0; it < NN / 16; ++it) {
        int cur = it & 1;
        // async prefetch of the next panel into the other buffer
        if (it + 1 < NN / 16) {
            unsigned long long g =
                (unsigned long long)(pbase + (size_t)(it + 1) * 16 * CIN);
#pragma unroll
            for (int i = 0; i < 4; ++i)
                async_b128(ldsb[1 - cur] + tid * 16 + i * 2048,
                           g + tid * 16 + i * 2048);
        }

        // B fragments from LDS panel (batched ds loads), then WMMA burst
        const float* pc = &panel[cur][0];
        v2f brg[32];
#pragma unroll
        for (int kk = 0; kk < 32; ++kk)
            brg[kk] = *(const v2f*)(pc + lm * CIN + 4 * kk + 2 * half);

        v8f acc0 = zero8(), acc1 = zero8();         // two independent chains
#pragma unroll
        for (int kk = 0; kk < 32; kk += 2) {
            acc0 = __builtin_amdgcn_wmma_f32_16x16x4_f32(
                false, a[kk], false, brg[kk], (short)0, acc0, false, false);
            acc1 = __builtin_amdgcn_wmma_f32_16x16x4_f32(
                false, a[kk + 1], false, brg[kk + 1], (short)0, acc1, false, false);
        }

        float sqb = sq[b * NN + it * 16 + lm];
#pragma unroll
        for (int r = 0; r < 8; ++r)
            tile[wave][(r + 8 * half) * 16 + lm] =
                sqa[r] + sqb - 2.0f * (acc0[r] + acc1[r]);

        // per-row top-K update (intra-wave; DS ops are in-order per wave)
        if (l < 16) {
            int colbase = it * 16;
#pragma unroll
            for (int j = 0; j < 16; ++j) {
                float d = tile[wave][l * 16 + j];
                if (d < wmax) {
                    int col = colbase + j;
#pragma unroll
                    for (int s = 0; s < 16; ++s)
                        if (s == wpos) { bd[s] = d; bi[s] = col; }
                    wmax = -3.0e38f;
#pragma unroll
                    for (int s = 0; s < 16; ++s)
                        if (bd[s] > wmax) { wmax = bd[s]; wpos = s; }
                }
            }
        }
        wait_async0();                               // next panel landed
        __syncthreads();
    }
    if (l < 16) {
        int* op = knn + ((size_t)(b * NN + rowbase + l)) * KNB;
#pragma unroll
        for (int s = 0; s < 16; ++s) op[s] = bi[s];
    }
}

// ---------------------------------------------------------------------------
// K2: edge MLP (f16 WMMA) + BN1 + ReLU + neighbor attention + aggregate.
//   One wave per query point: M=16 neighbors, K=256 (edge), N=128 (H).
//   Edge weights staged once per block into LDS (64 KB, 8 waves reuse).
// ---------------------------------------------------------------------------
__global__ __launch_bounds__(256) void k_edge(const float* __restrict__ xt,
                                              const int* __restrict__ knn,
                                              const _Float16* __restrict__ WeH,
                                              const float* __restrict__ s1,
                                              const float* __restrict__ b1,
                                              const float* __restrict__ w_att,
                                              _Float16* __restrict__ aggH) {
    __shared__ _Float16 smWe[HH * 2 * CIN];        // 32768 halves = 64 KB
    {
        const uint4* src = reinterpret_cast<const uint4*>(WeH);
        uint4* dst = reinterpret_cast<uint4*>(smWe);
        int t = threadIdx.x;
#pragma unroll
        for (int i = 0; i < 16; ++i) dst[t + 256 * i] = src[t + 256 * i];
    }
    __syncthreads();

    int wave = threadIdx.x >> 5;
    int q = blockIdx.x * 8 + wave;                  // global query row b*N+n
    int b = q >> 12;
    int l = threadIdx.x & 31, half = l >> 4, lm = l & 15;

    int g = knn[(size_t)q * KNB + lm];              // neighbor for row lm
    const float* pn = xt + ((size_t)(b * NN + g)) * CIN;
    const float* pc = xt + ((size_t)q) * CIN;

    v8f acc[8];
#pragma unroll
    for (int t = 0; t < 8; ++t) acc[t] = zero8();

    for (int ch = 0; ch < 8; ++ch) {
        int kk = ch * 32;
        // A fragment: lane row = lm, halves at K = kk+8*half+{0..7}, +16
        float e[16];
        if (ch < 4) {                               // neigh - center
            int o0 = kk + 8 * half;
#pragma unroll
            for (int j = 0; j < 8; ++j) e[j]     = pn[o0 + j]      - pc[o0 + j];
#pragma unroll
            for (int j = 0; j < 8; ++j) e[8 + j] = pn[o0 + 16 + j] - pc[o0 + 16 + j];
        } else {                                    // center
            int o0 = kk - CIN + 8 * half;
#pragma unroll
            for (int j = 0; j < 8; ++j) e[j]     = pc[o0 + j];
#pragma unroll
            for (int j = 0; j < 8; ++j) e[8 + j] = pc[o0 + 16 + j];
        }
        V16H av;
#pragma unroll
        for (int j = 0; j < 16; ++j) av.v[j] = (_Float16)e[j];

#pragma unroll
        for (int t = 0; t < 8; ++t) {               // 8 output col-tiles of 16
            int hch = t * 16 + lm;
            V16H bv;
            const uint4* wp =
                reinterpret_cast<const uint4*>(&smWe[hch * (2 * CIN) + kk + 8 * half]);
            bv.q[0] = wp[0];                        // K offsets +0..7
            bv.q[1] = wp[2];                        // K offsets +16..23
            acc[t] = __builtin_amdgcn_wmma_f32_16x16x32_f16(
                false, av.v, false, bv.v, (short)0, acc[t], false, false);
        }
    }

    // BN1 + ReLU + attention logits
    float pl[8];
#pragma unroll
    for (int r = 0; r < 8; ++r) pl[r] = 0.0f;
#pragma unroll
    for (int t = 0; t < 8; ++t) {
        int hch = t * 16 + lm;
        float s = s1[hch], bb = b1[hch], wa = w_att[hch];
#pragma unroll
        for (int r = 0; r < 8; ++r) {
            float hv = fmaxf(acc[t][r] * s + bb, 0.0f);
            acc[t][r] = hv;
            pl[r] += hv * wa;
        }
    }
    // reduce logits over the 16 lanes of each half
#pragma unroll
    for (int off = 1; off < 16; off <<= 1)
#pragma unroll
        for (int r = 0; r < 8; ++r) pl[r] += __shfl_xor(pl[r], off, 32);
    float ol[8];
#pragma unroll
    for (int r = 0; r < 8; ++r) ol[r] = __shfl_xor(pl[r], 16, 32);

    float mx = -3.0e38f;
#pragma unroll
    for (int r = 0; r < 8; ++r) mx = fmaxf(mx, fmaxf(pl[r], ol[r]));
    float ex[8], sum = 0.0f;
#pragma unroll
    for (int r = 0; r < 8; ++r) {
        ex[r] = __expf(pl[r] - mx);
        sum += ex[r] + __expf(ol[r] - mx);
    }
    float inv = 1.0f / sum;

    // aggregate: agg[h] = sum_k att[k] * h[k][h]
#pragma unroll
    for (int t = 0; t < 8; ++t) {
        float ap = 0.0f;
#pragma unroll
        for (int r = 0; r < 8; ++r) ap += ex[r] * inv * acc[t][r];
        ap += __shfl_xor(ap, 16, 32);               // add other half's k rows
        if (l < 16) aggH[(size_t)q * HH + t * 16 + l] = (_Float16)ap;
    }
}

// ---------------------------------------------------------------------------
// K3: fuse GEMM (agg x Wf^T) + BN2 + ReLU, residual GEMM (x x Wr^T) + BN3,
//     plus SE per-chunk channel sums via atomics. One wave per 16 rows.
// ---------------------------------------------------------------------------
__global__ __launch_bounds__(32) void k_fuse_res(
    const _Float16* __restrict__ aggH, const _Float16* __restrict__ xth,
    const _Float16* __restrict__ WfH,  const _Float16* __restrict__ WrH,
    const float* s2, const float* b2, const float* s3, const float* b3,
    float* __restrict__ tmp, float* __restrict__ res, float* __restrict__ ssum) {
    int q0 = blockIdx.x * 16;
    int b = q0 >> 12, n0 = q0 & (NN - 1), chunk = n0 >> 8;
    int l = threadIdx.x, half = l >> 4, lm = l & 15;

    const _Float16* ar = aggH + (size_t)(q0 + lm) * HH;
    const _Float16* xr = xth  + (size_t)(q0 + lm) * CIN;
    V16H fa[4], ra[4];
#pragma unroll
    for (int c = 0; c < 4; ++c) {
        int kk = c * 32;
        const uint4* p  = (const uint4*)(ar + kk + 8 * half);
        fa[c].q[0] = p[0];  fa[c].q[1] = p[2];
        const uint4* p2 = (const uint4*)(xr + kk + 8 * half);
        ra[c].q[0] = p2[0]; ra[c].q[1] = p2[2];
    }

    float* ssp = ssum + (size_t)(b * NC + chunk) * COUT;
    for (int t = 0; t < 16; ++t) {
        int o = t * 16 + lm;
        const _Float16* wf = WfH + (size_t)o * HH;
        const _Float16* wr = WrH + (size_t)o * CIN;
        // batch all B-fragment loads ahead of the WMMA burst
        V16H bf[4], br[4];
#pragma unroll
        for (int c = 0; c < 4; ++c) {
            int kk = c * 32;
            const uint4* pf = (const uint4*)(wf + kk + 8 * half);
            bf[c].q[0] = pf[0]; bf[c].q[1] = pf[2];
            const uint4* pr = (const uint4*)(wr + kk + 8 * half);
            br[c].q[0] = pr[0]; br[c].q[1] = pr[2];
        }
        v8f aF = zero8(), aR = zero8();
#pragma unroll
        for (int c = 0; c < 4; ++c) {
            aF = __builtin_amdgcn_wmma_f32_16x16x32_f16(
                false, fa[c].v, false, bf[c].v, (short)0, aF, false, false);
            aR = __builtin_amdgcn_wmma_f32_16x16x32_f16(
                false, ra[c].v, false, br[c].v, (short)0, aR, false, false);
        }
        float sf = s2[o], bf2 = b2[o], sr = s3[o], br3 = b3[o];
        float part = 0.0f;
#pragma unroll
        for (int r = 0; r < 8; ++r) {
            int m = r + 8 * half;
            float tv = fmaxf(aF[r] * sf + bf2, 0.0f);
            tmp[(size_t)(q0 + m) * COUT + o] = tv;
            part += tv;
            res[(size_t)(q0 + m) * COUT + o] = aR[r] * sr + br3;
        }
        part += __shfl_xor(part, 16, 32);
        if (l < 16) atomicAdd(&ssp[o], part);
    }
}

// ---------------------------------------------------------------------------
// K4: SE MLP per (b, chunk): mean -> fc1 -> relu -> fc2 -> sigmoid
// ---------------------------------------------------------------------------
__global__ __launch_bounds__(256) void k_se(const float* __restrict__ ssum,
                                            const float* f1w, const float* f1b,
                                            const float* f2w, const float* f2b,
                                            float* __restrict__ se) {
    __shared__ float sm[COUT];
    __shared__ float hb[SEH];
    int row = blockIdx.x, t = threadIdx.x;
    sm[t] = ssum[(size_t)row * COUT + t] * (1.0f / 256.0f);
    __syncthreads();
    if (t < SEH) {
        float a = f1b[t];
        for (int c = 0; c < COUT; ++c) a += sm[c] * f1w[t * COUT + c];
        hb[t] = fmaxf(a, 0.0f);
    }
    __syncthreads();
    float a = f2b[t];
#pragma unroll
    for (int j = 0; j < SEH; ++j) a += hb[j] * f2w[t * SEH + j];
    se[(size_t)row * COUT + t] = 1.0f / (1.0f + __expf(-a));
}

// ---------------------------------------------------------------------------
// K5: out(B,Cout,N) = tmp * se + res, transposed write
// ---------------------------------------------------------------------------
__global__ __launch_bounds__(256) void k_final(const float* __restrict__ tmp,
                                               const float* __restrict__ res,
                                               const float* __restrict__ se,
                                               float* __restrict__ out) {
    int id = blockIdx.x * 256 + threadIdx.x;        // ((b*Cout+o)*N+n)
    int n = id & (NN - 1);
    int o = (id >> 12) & (COUT - 1);
    int b = id >> 20;
    size_t ridx = ((size_t)(b * NN + n)) * COUT + o;
    float sev = se[((size_t)(b * NC + (n >> 8))) * COUT + o];
    out[id] = tmp[ridx] * sev + res[ridx];
}

// ---------------------------------------------------------------------------
extern "C" void kernel_launch(void* const* d_in, const int* in_sizes, int n_in,
                              void* d_out, int out_size, void* d_ws, size_t ws_size,
                              hipStream_t stream) {
    (void)in_sizes; (void)n_in; (void)out_size; (void)ws_size;
    const float* x      = (const float*)d_in[0];
    const float* W_edge = (const float*)d_in[1];
    const float* bn1_g  = (const float*)d_in[2];
    const float* bn1_b  = (const float*)d_in[3];
    const float* bn1_m  = (const float*)d_in[4];
    const float* bn1_v  = (const float*)d_in[5];
    const float* w_att  = (const float*)d_in[6];
    const float* W_fuse = (const float*)d_in[7];
    const float* bn2_g  = (const float*)d_in[8];
    const float* bn2_b  = (const float*)d_in[9];
    const float* bn2_m  = (const float*)d_in[10];
    const float* bn2_v  = (const float*)d_in[11];
    const float* fc1_w  = (const float*)d_in[12];
    const float* fc1_b  = (const float*)d_in[13];
    const float* fc2_w  = (const float*)d_in[14];
    const float* fc2_b  = (const float*)d_in[15];
    const float* W_res  = (const float*)d_in[16];
    const float* bn3_g  = (const float*)d_in[17];
    const float* bn3_b  = (const float*)d_in[18];
    const float* bn3_m  = (const float*)d_in[19];
    const float* bn3_v  = (const float*)d_in[20];
    float* out = (float*)d_out;

    char* w = (char*)d_ws;
    auto alloc = [&](size_t bytes) -> char* {
        char* p = w;
        w += (bytes + 255) & ~(size_t)255;
        return p;
    };
    float*    xt   = (float*)   alloc((size_t)BB * NN * CIN * 4);
    _Float16* xth  = (_Float16*)alloc((size_t)BB * NN * CIN * 2);
    float*    sq   = (float*)   alloc((size_t)BB * NN * 4);
    int*      knn  = (int*)     alloc((size_t)BB * NN * KNB * 4);
    _Float16* WeH  = (_Float16*)alloc((size_t)HH * 2 * CIN * 2);
    _Float16* WfH  = (_Float16*)alloc((size_t)COUT * HH * 2);
    _Float16* WrH  = (_Float16*)alloc((size_t)COUT * CIN * 2);
    float*    s1   = (float*)   alloc(HH * 4);
    float*    b1   = (float*)   alloc(HH * 4);
    float*    s2   = (float*)   alloc(COUT * 4);
    float*    b2   = (float*)   alloc(COUT * 4);
    float*    s3   = (float*)   alloc(COUT * 4);
    float*    b3   = (float*)   alloc(COUT * 4);
    _Float16* aggH = (_Float16*)alloc((size_t)BB * NN * HH * 2);
    float*    tmp  = (float*)   alloc((size_t)BB * NN * COUT * 4);
    float*    res  = (float*)   alloc((size_t)BB * NN * COUT * 4);
    float*    ssum = (float*)   alloc((size_t)BB * NC * COUT * 4);
    float*    se   = (float*)   alloc((size_t)BB * NC * COUT * 4);

    hipMemsetAsync(ssum, 0, (size_t)BB * NC * COUT * 4, stream);

    k_prep_x<<<(BB * NN) / 256, 256, 0, stream>>>(x, xt, xth, sq);
    k_prep_w<<<128, 256, 0, stream>>>(W_edge, W_fuse, W_res,
                                      bn1_g, bn1_b, bn1_m, bn1_v,
                                      bn2_g, bn2_b, bn2_m, bn2_v,
                                      bn3_g, bn3_b, bn3_m, bn3_v,
                                      WeH, WfH, WrH, s1, b1, s2, b2, s3, b3);
    k_knn<<<dim3(NN / 64, BB), 128, 0, stream>>>(xt, sq, knn);
    k_edge<<<(BB * NN) / 8, 256, 0, stream>>>(xt, knn, WeH, s1, b1, w_att, aggH);
    k_fuse_res<<<(BB * NN) / 16, 32, 0, stream>>>(aggH, xth, WfH, WrH,
                                                  s2, b2, s3, b3, tmp, res, ssum);
    k_se<<<BB * NC, 256, 0, stream>>>(ssum, fc1_w, fc1_b, fc2_w, fc2_b, se);
    k_final<<<(BB * COUT * NN) / 256, 256, 0, stream>>>(tmp, res, se, out);
}
